// Network_54288386621762
// MI455X (gfx1250) — compile-verified, matrix operand order checked by reading, not run on hardware
//
#include <hip/hip_runtime.h>
#include <hip/hip_bf16.h>

typedef __attribute__((ext_vector_type(16))) _Float16 v16h;
typedef __attribute__((ext_vector_type(8)))  float    v8f;

constexpr int B = 8, N = 8192, M = 512, K = 6, NS = 16, C = 13, D = 64;
constexpr size_t BN = (size_t)B * N;
constexpr size_t BM = (size_t)B * M;

// ---- d_out float offsets (tuple outputs concatenated flat, reference order) ----
constexpr size_t O_FINAL  = 0;                   // final_asso  [B,N,K]
constexpr size_t O_CIDX   = O_FINAL  + BN * K;   // cluster_idx [B,M] (as float)
constexpr size_t O_C2P    = O_CIDX   + BM;       // c2p_idx     [B,N,K] (as float)
constexpr size_t O_C2PABS = O_C2P    + BN * K;   // c2p_idx_abs [B,N,K] (as float)
constexpr size_t O_OUT    = O_C2PABS + BN * K;   // out         [B,N,C]
constexpr size_t O_REXYZ  = O_OUT    + BN * C;   // re_p_xyz    [B,N,3]
constexpr size_t O_RELAB  = O_REXYZ  + BN * 3;   // re_p_label  [B,N,C]
constexpr size_t O_FDIST  = O_RELAB  + BN * C;   // fea_dist    [B,N,K]
constexpr size_t O_PFEA   = O_FDIST  + BN * K;   // p_fea       [B,N,D]
constexpr size_t O_SPLABT = O_PFEA   + BN * D;   // sp_label^T  [B,C,M]
constexpr size_t O_PSEUDO = O_SPLABT + BM * C;   // sp_pseudo   [B,M] (as float)
constexpr size_t O_PSONE  = O_PSEUDO + BM;       // pseudo 1hot [B,C,M]

// =========================== WMMA helpers (wave32) ===========================
__device__ __forceinline__ int lane_id() { return (int)(threadIdx.x & 31u); }

__device__ __forceinline__ void lds_fence() {
  asm volatile("s_wait_dscnt 0" ::: "memory");   // CDNA5 split DS counter
}

__device__ __forceinline__ v8f wmma_f16f32(v16h a, v16h b, v8f c) {
  // D = A(16x32 f16) x B(32x16 f16) + C(16x16 f32)
  return __builtin_amdgcn_wmma_f32_16x16x32_f16(false, a, false, b, (short)0, c,
                                                false, false);
}

// K index of half-pair slot v (0..7) for lane group g (ISA 16-bit A layout)
__device__ __forceinline__ int wmma_kslot(int v, int g) {
  return ((v < 4) ? 0 : 16) + g * 8 + ((v & 3) << 1);
}

__device__ __forceinline__ v16h load_A_lds_f16(const _Float16* t, int ld, int k0) {
  int l = lane_id(), g = l >> 4, row = l & 15;
  v16h a;
#pragma unroll
  for (int v = 0; v < 8; ++v) {
    int kb = k0 + wmma_kslot(v, g);
    a[2 * v]     = t[row * ld + kb];
    a[2 * v + 1] = t[row * ld + kb + 1];
  }
  return a;
}

__device__ __forceinline__ v16h load_A_lds_f32(const float* t, int ld, int k0, int kvalid) {
  int l = lane_id(), g = l >> 4, row = l & 15;
  v16h a;
#pragma unroll
  for (int v = 0; v < 8; ++v) {
    int kb = k0 + wmma_kslot(v, g);
    float x0 = (kb     < kvalid) ? t[row * ld + kb]     : 0.0f;
    float x1 = (kb + 1 < kvalid) ? t[row * ld + kb + 1] : 0.0f;
    a[2 * v]     = (_Float16)x0;
    a[2 * v + 1] = (_Float16)x1;
  }
  return a;
}

__device__ __forceinline__ v16h load_A_gl_f16(const _Float16* rows, int ld, int k0) {
  int l = lane_id(), g = l >> 4, row = l & 15;
  v16h a;
#pragma unroll
  for (int v = 0; v < 8; ++v) {
    int kb = k0 + wmma_kslot(v, g);
    a[2 * v]     = rows[(size_t)row * ld + kb];
    a[2 * v + 1] = rows[(size_t)row * ld + kb + 1];
  }
  return a;
}

// B matrix (32x16, K x N) from row-major f32 W[ktotal x ldn], cols col0..+15
__device__ __forceinline__ v16h load_B_gl_f32(const float* Wm, int ldn, int ncols,
                                              int col0, int k0, int kvalid) {
  int l = lane_id(), g = l >> 4, col = col0 + (l & 15);
  v16h bb;
#pragma unroll
  for (int v = 0; v < 8; ++v) {
    int kb = k0 + wmma_kslot(v, g);
    float x0 = (kb     < kvalid && col < ncols) ? Wm[(size_t)kb * ldn + col]       : 0.0f;
    float x1 = (kb + 1 < kvalid && col < ncols) ? Wm[(size_t)(kb + 1) * ldn + col] : 0.0f;
    bb[2 * v]     = (_Float16)x0;
    bb[2 * v + 1] = (_Float16)x1;
  }
  return bb;
}

// ================================ kernels ===================================
__global__ void k_zero(float* p, int n) {
  int i = blockIdx.x * 256 + threadIdx.x;
  if (i < n) p[i] = 0.0f;
}

// ---- furthest point sampling: one block per batch, sequential over M steps ----
__global__ void __launch_bounds__(1024) k_fps(const float* __restrict__ pc,
                                              int* __restrict__ cidx,
                                              float* __restrict__ cxyz,
                                              float* __restrict__ out) {
  __shared__ float dist[N];            // 32 KB of the 320 KB WGP LDS
  __shared__ float sv[32];
  __shared__ int   si[32];
  __shared__ float s_last[3];
  __shared__ int   s_lasti;
  const int b = blockIdx.x, t = threadIdx.x;
  const float* base = pc + (size_t)b * N * 6;
  for (int i = t; i < N; i += 1024) dist[i] = 1e10f;
  if (t == 0) { s_lasti = 0; s_last[0] = base[0]; s_last[1] = base[1]; s_last[2] = base[2]; }
  __syncthreads();
  for (int m = 0; m < M; ++m) {
    if (t == 0) {
      int last = s_lasti;
      cidx[b * M + m] = last;
      out[O_CIDX + b * M + m] = (float)last;
      float* cx = cxyz + ((size_t)b * M + m) * 3;
      cx[0] = s_last[0]; cx[1] = s_last[1]; cx[2] = s_last[2];
    }
    const float lx = s_last[0], ly = s_last[1], lz = s_last[2];
    float bv = -1.0f; int bi = 0x7fffffff;
    for (int i = t; i < N; i += 1024) {
      const float* q = base + (size_t)i * 6;
      float dx = q[0] - lx, dy = q[1] - ly, dz = q[2] - lz;
      float d  = dx * dx + dy * dy + dz * dz;
      float nd = fminf(dist[i], d);
      dist[i] = nd;
      if (nd > bv || (nd == bv && i < bi)) { bv = nd; bi = i; }
    }
#pragma unroll
    for (int off = 16; off; off >>= 1) {               // wave32 butterfly argmax
      float ov = __shfl_xor(bv, off);
      int   oi = __shfl_xor(bi, off);
      if (ov > bv || (ov == bv && oi < bi)) { bv = ov; bi = oi; }
    }
    if ((t & 31) == 0) { sv[t >> 5] = bv; si[t >> 5] = bi; }
    __syncthreads();
    if (t < 32) {
      bv = sv[t]; bi = si[t];
#pragma unroll
      for (int off = 16; off; off >>= 1) {
        float ov = __shfl_xor(bv, off);
        int   oi = __shfl_xor(bi, off);
        if (ov > bv || (ov == bv && oi < bi)) { bv = ov; bi = oi; }
      }
      if (t == 0) {
        s_lasti = bi;
        const float* q = base + (size_t)bi * 6;
        s_last[0] = q[0]; s_last[1] = q[1]; s_last[2] = q[2];
      }
    }
    __syncthreads();
  }
}

// ---- K nearest cluster centers per point (centers staged in LDS) ----
__global__ void k_knn(const float* __restrict__ pc, const float* __restrict__ cxyz,
                      const int* __restrict__ cidx, int* __restrict__ c2pabs,
                      float* __restrict__ out) {
  __shared__ float cx[M * 3];          // 6 KB
  const int bpb = N / 256;
  const int b = blockIdx.x / bpb;
  const int n = (blockIdx.x % bpb) * 256 + threadIdx.x;
  for (int i = threadIdx.x; i < M * 3; i += 256) cx[i] = cxyz[(size_t)b * M * 3 + i];
  __syncthreads();
  const float* q = pc + ((size_t)b * N + n) * 6;
  const float px = q[0], py = q[1], pz = q[2];
  float bd[K]; int bi[K];
#pragma unroll
  for (int k = 0; k < K; ++k) { bd[k] = 3.4e38f; bi[k] = 0; }
  for (int m = 0; m < M; ++m) {
    float dx = cx[m * 3] - px, dy = cx[m * 3 + 1] - py, dz = cx[m * 3 + 2] - pz;
    float d = dx * dx + dy * dy + dz * dz;
    if (d < bd[K - 1]) {
      int pos = K - 1;
      while (pos > 0 && d < bd[pos - 1]) { bd[pos] = bd[pos - 1]; bi[pos] = bi[pos - 1]; --pos; }
      bd[pos] = d; bi[pos] = m;
    }
  }
  size_t o = ((size_t)b * N + n) * K;
#pragma unroll
  for (int k = 0; k < K; ++k) {
    c2pabs[o + k] = bi[k];
    out[O_C2PABS + o + k] = (float)bi[k];
    out[O_C2P    + o + k] = (float)cidx[b * M + bi[k]];
  }
}

// ---- p_fea = max_NS(mlp2(knn)) + relu(fea@Wg) : one wave per point (WMMA) ----
__global__ void __launch_bounds__(128) k_pfea(const float* __restrict__ pc,
                                              const float* __restrict__ knn,
                                              const float* __restrict__ Wk1,
                                              const float* __restrict__ Wk2,
                                              const float* __restrict__ Wg,
                                              float* __restrict__ pfea,
                                              _Float16* __restrict__ pfeah,
                                              float* __restrict__ out_pfea) {
  __shared__ _Float16 s_h1[4][16 * 32];   // per-wave layer-1 staging (4 KB)
  const int wv = threadIdx.x >> 5, lane = lane_id();
  const int p = blockIdx.x * 4 + wv;      // global point, exact cover of B*N
  const int b = p / N, n = p % N;
  v16h Bt[4];
#pragma unroll
  for (int t = 0; t < 4; ++t) Bt[t] = load_B_gl_f32(Wk2, D, D, t * 16, 0, 32);
  if (lane < 16) {                         // layer 1: K=2, VALU, NS=16 rows
    const float* xr = knn + (((size_t)b * N + n) * NS + lane) * 2;
    const float x0 = xr[0], x1 = xr[1];
#pragma unroll
    for (int c = 0; c < 32; ++c) {
      float h = x0 * Wk1[c] + x1 * Wk1[32 + c];
      s_h1[wv][lane * 32 + c] = (_Float16)fmaxf(h, 0.0f);
    }
  }
  lds_fence();
  v16h A = load_A_lds_f16(s_h1[wv], 32, 0);
  float hk4[4];
#pragma unroll
  for (int t = 0; t < 4; ++t) {
    v8f c = {};
    c = wmma_f16f32(A, Bt[t], c);
    float m8 = c[0];
#pragma unroll
    for (int r = 1; r < 8; ++r) m8 = fmaxf(m8, c[r]);
    m8 = fmaxf(m8, __shfl_xor(m8, 16));    // combine row halves (C layout)
    hk4[t] = fmaxf(m8, 0.0f);              // relu(max) == max(relu)
  }
  const float* fr = pc + ((size_t)b * N + n) * 6 + 3;
  const float fx = fr[0], fy = fr[1], fz = fr[2];
  if (lane < 16) {
#pragma unroll
    for (int t = 0; t < 4; ++t) {
      int d = t * 16 + lane;
      float hg = fmaxf(fx * Wg[d] + fy * Wg[64 + d] + fz * Wg[128 + d], 0.0f);
      float v  = hk4[t] + hg;
      size_t o = ((size_t)b * N + n) * D + d;
      pfea[o] = v; pfeah[o] = (_Float16)v; out_pfea[o] = v;
    }
  }
}

// ---- head: out = relu(p_fea@W1)@W2 : one wave per 16-row tile (WMMA) ----
__global__ void __launch_bounds__(128) k_out(const _Float16* __restrict__ pfeah,
                                             const float* __restrict__ W1,
                                             const float* __restrict__ W2,
                                             float* __restrict__ out) {
  __shared__ _Float16 s_hid[4][16 * 64];   // 8 KB
  const int wv = threadIdx.x >> 5, lane = lane_id();
  const int tile = blockIdx.x * 4 + wv;    // B*N/16 tiles exactly
  const _Float16* Arow = pfeah + (size_t)tile * 16 * D;
  __builtin_prefetch(Arow + 16 * D, 0, 1); // global_prefetch_b8 next tile
  v16h A0 = load_A_gl_f16(Arow, D, 0);
  v16h A1 = load_A_gl_f16(Arow, D, 32);
  const int col = lane & 15, rb = (lane >> 4) * 8;
#pragma unroll
  for (int nt = 0; nt < 4; ++nt) {
    v16h B0 = load_B_gl_f32(W1, D, D, nt * 16, 0, D);
    v16h B1 = load_B_gl_f32(W1, D, D, nt * 16, 32, D);
    v8f c = {};
    c = wmma_f16f32(A0, B0, c);
    c = wmma_f16f32(A1, B1, c);
#pragma unroll
    for (int r = 0; r < 8; ++r)
      s_hid[wv][(rb + r) * 64 + nt * 16 + col] = (_Float16)fmaxf(c[r], 0.0f);
  }
  lds_fence();
  v16h H0 = load_A_lds_f16(s_hid[wv], 64, 0);
  v16h H1 = load_A_lds_f16(s_hid[wv], 64, 32);
  v16h B20 = load_B_gl_f32(W2, C, C, 0, 0, D);
  v16h B21 = load_B_gl_f32(W2, C, C, 0, 32, D);
  v8f c2 = {};
  c2 = wmma_f16f32(H0, B20, c2);
  c2 = wmma_f16f32(H1, B21, c2);
  if (col < C) {
#pragma unroll
    for (int r = 0; r < 8; ++r)
      out[((size_t)tile * 16 + rb + r) * C + col] = c2[r];
  }
}

// ---- pm = mlp2(p_fea, Wm1, Wm2) (64->16->16, WMMA) ----
__global__ void __launch_bounds__(128) k_pm(const _Float16* __restrict__ pfeah,
                                            const float* __restrict__ Wm1,
                                            const float* __restrict__ Wm2,
                                            float* __restrict__ pmbuf) {
  __shared__ float s_hid[4][16 * 16];      // 4 KB
  const int wv = threadIdx.x >> 5, lane = lane_id();
  const int tile = blockIdx.x * 4 + wv;
  const _Float16* Arow = pfeah + (size_t)tile * 16 * D;
  v16h A0 = load_A_gl_f16(Arow, D, 0);
  v16h A1 = load_A_gl_f16(Arow, D, 32);
  v16h B0 = load_B_gl_f32(Wm1, 16, 16, 0, 0, D);
  v16h B1 = load_B_gl_f32(Wm1, 16, 16, 0, 32, D);
  v8f c = {};
  c = wmma_f16f32(A0, B0, c);
  c = wmma_f16f32(A1, B1, c);
  const int col = lane & 15, rb = (lane >> 4) * 8;
#pragma unroll
  for (int r = 0; r < 8; ++r) s_hid[wv][(rb + r) * 16 + col] = fmaxf(c[r], 0.0f);
  lds_fence();
  v16h H  = load_A_lds_f32(s_hid[wv], 16, 0, 16);
  v16h B2 = load_B_gl_f32(Wm2, 16, 16, 0, 0, 16);
  v8f c2 = {};
  c2 = wmma_f16f32(H, B2, c2);
#pragma unroll
  for (int r = 0; r < 8; ++r)
    pmbuf[((size_t)tile * 16 + rb + r) * 16 + col] = fmaxf(c2[r], 0.0f);
}

// ---- SLIC logits: one wave per 16 (b,n,k) rows; wf & wx via WMMA ----
__global__ void __launch_bounds__(128) k_logits(const float* __restrict__ pfea,
                                                const float* __restrict__ spfea,
                                                const float* __restrict__ spxyz,
                                                const float* __restrict__ pc,
                                                const int* __restrict__ c2pabs,
                                                const float* __restrict__ pmbuf,
                                                const float* __restrict__ Wf1,
                                                const float* __restrict__ Wf2,
                                                const float* __restrict__ Wx1,
                                                const float* __restrict__ Wx2,
                                                float* __restrict__ logits) {
  __shared__ float s_fd[4][16 * 64];       // 16 KB
  __shared__ float s_xd[4][16 * 4];        // 1 KB
  __shared__ float s_pm[4][16 * 16];       // 4 KB
  __shared__ float s_hid[4][16 * 16];      // 4 KB
  const int wv = threadIdx.x >> 5, lane = lane_id();
  const int tile = blockIdx.x * 4 + wv;    // B*N*K/16 tiles exactly
  {                                        // cooperative staging: 2 lanes per row
    const int r = lane & 15, half = lane >> 4;
    const size_t row = (size_t)tile * 16 + r;
    const size_t pnt = row / K;            // flattened (b,n)
    const int b = (int)(pnt / N);
    const int m = c2pabs[row];
    const float* sf = spfea + ((size_t)b * M + m) * D;
    const float* pf = pfea + pnt * D;
    for (int c = half * 32; c < half * 32 + 32; ++c) s_fd[wv][r * 64 + c] = sf[c] - pf[c];
    if (half == 0) {
      const float* sx = spxyz + ((size_t)b * M + m) * 3;
      const float* px = pc + pnt * 6;
#pragma unroll
      for (int a = 0; a < 3; ++a) s_xd[wv][r * 4 + a] = sx[a] - px[a];
      s_xd[wv][r * 4 + 3] = 0.0f;
#pragma unroll
      for (int j = 0; j < 16; ++j) s_pm[wv][r * 16 + j] = pmbuf[pnt * 16 + j];
    }
  }
  lds_fence();
  const int col = lane & 15, rb = (lane >> 4) * 8;
  // wf = mlp2(fd, 64->16->16)
  v16h A0 = load_A_lds_f32(s_fd[wv], 64, 0, D);
  v16h A1 = load_A_lds_f32(s_fd[wv], 64, 32, D);
  {
    v16h B0 = load_B_gl_f32(Wf1, 16, 16, 0, 0, D);
    v16h B1 = load_B_gl_f32(Wf1, 16, 16, 0, 32, D);
    v8f c = {};
    c = wmma_f16f32(A0, B0, c);
    c = wmma_f16f32(A1, B1, c);
#pragma unroll
    for (int r = 0; r < 8; ++r) s_hid[wv][(rb + r) * 16 + col] = fmaxf(c[r], 0.0f);
  }
  lds_fence();
  v16h Hf = load_A_lds_f32(s_hid[wv], 16, 0, 16);
  v16h Bf2 = load_B_gl_f32(Wf2, 16, 16, 0, 0, 16);
  v8f WF = {};
  WF = wmma_f16f32(Hf, Bf2, WF);
  // wx = mlp2(xd, 3->16->16), K padded with zeros
  v16h Ax  = load_A_lds_f32(s_xd[wv], 4, 0, 3);
  v16h Bx1 = load_B_gl_f32(Wx1, 16, 16, 0, 0, 3);
  {
    v8f cx = {};
    cx = wmma_f16f32(Ax, Bx1, cx);
#pragma unroll
    for (int r = 0; r < 8; ++r) s_hid[wv][(rb + r) * 16 + col] = fmaxf(cx[r], 0.0f);
  }
  lds_fence();
  v16h Hx  = load_A_lds_f32(s_hid[wv], 16, 0, 16);
  v16h Bx2 = load_B_gl_f32(Wx2, 16, 16, 0, 0, 16);
  v8f WX = {};
  WX = wmma_f16f32(Hx, Bx2, WX);
  // logits[row] = sum_j wf*wx*pm ; C layout -> 16-lane row reductions
#pragma unroll
  for (int r = 0; r < 8; ++r) {
    float wf = fmaxf(WF[r], 0.0f);
    float wx = fmaxf(WX[r], 0.0f);
    float pm = s_pm[wv][(rb + r) * 16 + col];
    float v = wf * wx * pm;
#pragma unroll
    for (int off = 1; off < 16; off <<= 1) v += __shfl_xor(v, off);
    if (col == 0) logits[(size_t)tile * 16 + rb + r] = v;
  }
}

// ---- initial superpoint mean scatter (counts + labels) ----
__global__ void k_init_scatter(const int* __restrict__ c2pabs, const int* __restrict__ label,
                               const float* __restrict__ pfea, float* __restrict__ numfea,
                               float* __restrict__ den, float* __restrict__ splab) {
  const int p = blockIdx.x * 256 + threadIdx.x;
  const int b = p / N;
  const int lab = label[p];
#pragma unroll
  for (int k = 0; k < K; ++k) {
    int m = c2pabs[(size_t)p * K + k];
    size_t o = (size_t)b * M + m;
    float* nf = numfea + o * D;
    const float* pf = pfea + (size_t)p * D;
    for (int d = 0; d < D; ++d) atomicAdd(nf + d, pf[d]);
    atomicAdd(den + o, 1.0f);
    atomicAdd(splab + o * C + lab, 1.0f);
  }
}

__global__ void k_norm_init(const float* __restrict__ numfea, const float* __restrict__ den,
                            float* __restrict__ spfea, const float* __restrict__ cxyz,
                            float* __restrict__ spxyz) {
  const int i = blockIdx.x * 256 + threadIdx.x;   // B*M
  const float dnz = 1.0f / fmaxf(den[i], 1.0f);
  for (int c = 0; c < D; ++c) spfea[(size_t)i * D + c] = numfea[(size_t)i * D + c] * dnz;
#pragma unroll
  for (int a = 0; a < 3; ++a) spxyz[(size_t)i * 3 + a] = cxyz[(size_t)i * 3 + a];
}

// ---- softmax over K + weighted scatter ----
__global__ void k_sm_scatter(float* __restrict__ wbuf, const int* __restrict__ c2pabs,
                             const float* __restrict__ pfea, const float* __restrict__ pc,
                             const float* __restrict__ onehot, int last,
                             float* __restrict__ numfea, float* __restrict__ numxyz,
                             float* __restrict__ den, float* __restrict__ numlab) {
  const int p = blockIdx.x * 256 + threadIdx.x;   // B*N
  const int b = p / N;
  float lg[K];
  float mx = -3.4e38f;
#pragma unroll
  for (int k = 0; k < K; ++k) { lg[k] = wbuf[(size_t)p * K + k]; mx = fmaxf(mx, lg[k]); }
  float s = 0.0f;
#pragma unroll
  for (int k = 0; k < K; ++k) { lg[k] = __expf(lg[k] - mx); s += lg[k]; }
  const float inv = 1.0f / s;
  const float* px = pc + (size_t)p * 6;
#pragma unroll
  for (int k = 0; k < K; ++k) {
    float w = lg[k] * inv;
    wbuf[(size_t)p * K + k] = w;
    int m = c2pabs[(size_t)p * K + k];
    size_t o = (size_t)b * M + m;
    atomicAdd(den + o, w);
    const float* pf = pfea + (size_t)p * D;
    for (int d = 0; d < D; ++d) atomicAdd(numfea + o * D + d, w * pf[d]);
#pragma unroll
    for (int a = 0; a < 3; ++a) atomicAdd(numxyz + o * 3 + a, w * px[a]);
    if (last)
      for (int c = 0; c < C; ++c)
        atomicAdd(numlab + o * C + c, w * onehot[(size_t)p * C + c]);
  }
}

__global__ void k_norm(const float* __restrict__ numfea, const float* __restrict__ numxyz,
                       const float* __restrict__ den, const float* __restrict__ numlab,
                       int last, float* __restrict__ spfea, float* __restrict__ spxyz,
                       float* __restrict__ splabel) {
  const int i = blockIdx.x * 256 + threadIdx.x;   // B*M
  const float dnz = 1.0f / fmaxf(den[i], 1e-8f);
  for (int c = 0; c < D; ++c) spfea[(size_t)i * D + c] = numfea[(size_t)i * D + c] * dnz;
#pragma unroll
  for (int a = 0; a < 3; ++a) spxyz[(size_t)i * 3 + a] = numxyz[(size_t)i * 3 + a] * dnz;
  if (last)
    for (int c = 0; c < C; ++c) splabel[(size_t)i * C + c] = numlab[(size_t)i * C + c] * dnz;
}

// ---- per-point final outputs ----
__global__ void k_final(const float* __restrict__ wbuf, const int* __restrict__ c2pabs,
                        const float* __restrict__ spxyz, const float* __restrict__ splabel,
                        float* __restrict__ out) {
  const int p = blockIdx.x * 256 + threadIdx.x;   // B*N
  const int b = p / N;
  float w[K];
  int bi = 0; float bv = -3.4e38f;
#pragma unroll
  for (int k = 0; k < K; ++k) {
    w[k] = wbuf[(size_t)p * K + k];
    out[O_FINAL + (size_t)p * K + k] = w[k];
    out[O_FDIST + (size_t)p * K + k] = w[k];
    if (w[k] > bv) { bv = w[k]; bi = k; }
  }
  const int sel = c2pabs[(size_t)p * K + bi];
#pragma unroll
  for (int a = 0; a < 3; ++a)
    out[O_REXYZ + (size_t)p * 3 + a] = spxyz[((size_t)b * M + sel) * 3 + a];
  float acc[C];
#pragma unroll
  for (int c = 0; c < C; ++c) acc[c] = 0.0f;
#pragma unroll
  for (int k = 0; k < K; ++k) {
    int m = c2pabs[(size_t)p * K + k];
    const float* sl = splabel + ((size_t)b * M + m) * C;
    for (int c = 0; c < C; ++c) acc[c] += sl[c] * w[k];
  }
  for (int c = 0; c < C; ++c) out[O_RELAB + (size_t)p * C + c] = acc[c];
}

// ---- per-superpoint final outputs (transpose + pseudo labels) ----
__global__ void k_final_sp(const float* __restrict__ splabel, const float* __restrict__ splab,
                           float* __restrict__ out) {
  const int i = blockIdx.x * 256 + threadIdx.x;   // B*M
  const int b = i / M, m = i % M;
  int bi = 0; float bv = -3.4e38f;
  for (int c = 0; c < C; ++c) {
    out[O_SPLABT + ((size_t)b * C + c) * M + m] = splabel[(size_t)i * C + c];
    float v = splab[(size_t)i * C + c];
    if (v > bv) { bv = v; bi = c; }
  }
  out[O_PSEUDO + i] = (float)bi;
  for (int c = 0; c < C; ++c)
    out[O_PSONE + ((size_t)b * C + c) * M + m] = (c == bi) ? 1.0f : 0.0f;
}

// ================================ launcher ==================================
extern "C" void kernel_launch(void* const* d_in, const int* in_sizes, int n_in,
                              void* d_out, int out_size, void* d_ws, size_t ws_size,
                              hipStream_t stream) {
  const float* pc     = (const float*)d_in[0];
  const float* knn    = (const float*)d_in[1];
  const float* onehot = (const float*)d_in[2];
  const int*   label  = (const int*)d_in[3];
  const float* Wk1    = (const float*)d_in[4];
  const float* Wk2    = (const float*)d_in[5];
  const float* Wg     = (const float*)d_in[6];
  const float* hW1    = (const float*)d_in[7];
  const float* hW2    = (const float*)d_in[8];
  const float* Wf1    = (const float*)d_in[9];   // [4,64,16]
  const float* Wf2    = (const float*)d_in[10];  // [4,16,16]
  const float* Wx1    = (const float*)d_in[11];  // [4,3,16]
  const float* Wx2    = (const float*)d_in[12];  // [4,16,16]
  const float* Wm1    = (const float*)d_in[13];  // [4,64,16]
  const float* Wm2    = (const float*)d_in[14];  // [4,16,16]
  float* out = (float*)d_out;

  // ---- workspace carve ----
  float*    cxyz    = (float*)d_ws;                         // BM*3
  int*      cidx    = (int*)(cxyz + BM * 3);                // BM
  int*      c2pabs  = (int*)(cidx + BM);                    // BN*K
  float*    pfea    = (float*)(c2pabs + BN * K);            // BN*D
  _Float16* pfeah   = (_Float16*)(pfea + BN * D);           // BN*D halfs
  float*    spfea   = (float*)(pfeah + BN * D);             // BM*D
  float*    spxyz   = spfea + BM * D;                       // BM*3
  float*    numfea  = spxyz + BM * 3;                       // BM*D   (accum block ->)
  float*    numxyz  = numfea + BM * D;                      // BM*3
  float*    den     = numxyz + BM * 3;                      // BM
  float*    numlab  = den + BM;                             // BM*C
  float*    splab   = numlab + BM * C;                      // BM*C   (<- accum block)
  float*    splabel = splab + BM * C;                       // BM*C
  float*    pmbuf   = splabel + BM * C;                     // BN*16
  float*    wbuf    = pmbuf + BN * 16;                      // BN*K
  (void)in_sizes; (void)n_in; (void)out_size; (void)ws_size;

  const int initZero  = (int)(BM * (D + 3 + 1 + C + C));    // numfea..splab
  const int layerZero = (int)(BM * (D + 3 + 1 + C));        // numfea..numlab

  k_zero<<<(initZero + 255) / 256, 256, 0, stream>>>(numfea, initZero);
  k_fps<<<B, 1024, 0, stream>>>(pc, cidx, cxyz, out);
  k_knn<<<B * N / 256, 256, 0, stream>>>(pc, cxyz, cidx, c2pabs, out);
  k_pfea<<<B * N / 4, 128, 0, stream>>>(pc, knn, Wk1, Wk2, Wg, pfea, pfeah, out + O_PFEA);
  k_out<<<B * N / 64, 128, 0, stream>>>(pfeah, hW1, hW2, out + O_OUT);
  k_init_scatter<<<B * N / 256, 256, 0, stream>>>(c2pabs, label, pfea, numfea, den, splab);
  k_norm_init<<<B * M / 256, 256, 0, stream>>>(numfea, den, spfea, cxyz, spxyz);

  for (int l = 0; l < 4; ++l) {
    k_pm<<<B * N / 64, 128, 0, stream>>>(pfeah, Wm1 + l * 64 * 16, Wm2 + l * 256, pmbuf);
    k_logits<<<B * N * K / 64, 128, 0, stream>>>(pfea, spfea, spxyz, pc, c2pabs, pmbuf,
                                                 Wf1 + l * 64 * 16, Wf2 + l * 256,
                                                 Wx1 + l * 48, Wx2 + l * 256, wbuf);
    k_zero<<<(layerZero + 255) / 256, 256, 0, stream>>>(numfea, layerZero);
    k_sm_scatter<<<B * N / 256, 256, 0, stream>>>(wbuf, c2pabs, pfea, pc, onehot,
                                                  (l == 3) ? 1 : 0,
                                                  numfea, numxyz, den, numlab);
    k_norm<<<B * M / 256, 256, 0, stream>>>(numfea, numxyz, den, numlab,
                                            (l == 3) ? 1 : 0, spfea, spxyz, splabel);
  }

  k_final<<<B * N / 256, 256, 0, stream>>>(wbuf, c2pabs, spxyz, splabel, out);
  k_final_sp<<<B * M / 256, 256, 0, stream>>>(splabel, splab, out);
}
